// Vision_Block_11304353923682
// MI455X (gfx1250) — matrix operand order repaired and checked by reading, not compiled, not run on hardware
//
#include <hip/hip_runtime.h>
#include <stdint.h>

typedef unsigned short bfraw;
typedef __attribute__((ext_vector_type(16))) __bf16 v16bf;
typedef __attribute__((ext_vector_type(8)))  float  v8f;

#define BN 4
#define CC 256
#define HH 64
#define WW 64
#define HWSZ (HH*WW)      /* 4096  */
#define NTOK (BN*HWSZ)    /* 16384 */
#define NHEADS 4
#define HIDC 680
#define DIC 512
#define DSC 16
#define DRC 16

// ---------------------------------------------------------------- helpers
__device__ __forceinline__ float b2f(bfraw h) {
  union { unsigned u; float f; } v; v.u = ((unsigned)h) << 16; return v.f;
}
__device__ __forceinline__ bfraw f2b(float f) {
  union { float f; unsigned u; } v; v.f = f;
  unsigned r = v.u + 0x7FFFu + ((v.u >> 16) & 1u);
  return (bfraw)(r >> 16);
}
__device__ __forceinline__ float act_apply(float v, int act, float p) {
  switch (act) {
    case 1: return v > 0.f ? v : v * p;                       // leaky relu
    case 2: return v > 0.f ? v : 0.f;                         // relu
    case 3: return 1.f / (1.f + __expf(-v));                  // sigmoid
    case 4: return v / (1.f + __expf(-v));                    // silu
    case 5: return 0.5f * v * (1.f + erff(v * 0.70710678f));  // gelu (exact)
    case 6: return v > 20.f ? v : log1pf(__expf(v));          // softplus
    default: return v;
  }
}
// per-lane async global->LDS 16B transfer (CDNA5 GLOBAL_LOAD_ASYNC_TO_LDS_B128).
// LDS destination address = low 32 bits of the generic pointer (ISA 10.2).
__device__ __forceinline__ void async_g2l_b128(const void* gp, void* lp) {
  unsigned lds = (unsigned)(uintptr_t)lp;
  asm volatile("global_load_async_to_lds_b128 %0, %1, off"
               :: "v"(lds), "v"(gp) : "memory");
}
__device__ __forceinline__ void async_wait0() {
  asm volatile("s_wait_asynccnt 0x0" ::: "memory");
}

// ---------------------------------------------------------------- WMMA GEMM
// Y[M,N] = act(A[M,K] * W[N,K]^T + bias), bf16 in / f32 accum / bf16 out.
// Preconditions (arranged by host): K % 64 == 0; W packed [Npad x Kw] with
// Npad covering every N-tile; A rows readable up to the 128-row tile (store
// is M-guarded). Staging is async (ASYNCcnt) and double-buffered so tile k+1
// streams into LDS while tile k is consumed by WMMA.
#define GT_M 128
#define GT_N 64
#define GT_K 64
#define LDS_S 72   /* 64 cols + 8 pad; row stride 144B (16B aligned) */

__global__ __launch_bounds__(256) void gemm_bf16_wmma(
    const bfraw* __restrict__ A, const bfraw* __restrict__ W,
    const float* __restrict__ bias, bfraw* __restrict__ Y,
    int M, int N, int K, int ldA, int Kw, int ldY, int act, float actp,
    long strideA, long strideW, long strideY)
{
  __shared__ __align__(16) bfraw As[2][GT_M][LDS_S];
  __shared__ __align__(16) bfraw Bs[2][GT_N][LDS_S];

  const int bz = blockIdx.z;
  A += strideA * bz; W += strideW * bz; Y += strideY * bz;

  const int tm   = blockIdx.y * GT_M;
  const int tn   = blockIdx.x * GT_N;
  const int tid  = threadIdx.x;
  const int lane = tid & 31;
  const int wave = tid >> 5;
  const int mf   = lane & 15;   // M (or N) index inside fragment
  const int sel  = lane >> 4;   // half-wave select

  v8f acc0 = {0,0,0,0,0,0,0,0};
  v8f acc1 = acc0, acc2 = acc0, acc3 = acc0;

  // issue async transfers for one K-tile into LDS buffer `buf`
  auto stage = [&](int buf, int k0) {
#pragma unroll
    for (int j = 0; j < 4; ++j) {           // A: 128x64 = 1024 x 16B, 4/thread
      const int c   = tid + j * 256;
      const int row = c >> 3;
      const int cb  = (c & 7) * 8;
      async_g2l_b128(A + (long)(tm + row) * ldA + k0 + cb, &As[buf][row][cb]);
    }
#pragma unroll
    for (int j = 0; j < 2; ++j) {           // B: 64x64 = 512 x 16B, 2/thread
      const int c   = tid + j * 256;
      const int row = c >> 3;
      const int cb  = (c & 7) * 8;
      async_g2l_b128(W + (long)(tn + row) * Kw + k0 + cb, &Bs[buf][row][cb]);
    }
  };

  stage(0, 0);
  int cur = 0;
  for (int k0 = 0; k0 < K; k0 += GT_K) {
    async_wait0();          // this wave's transfers for tile k0 landed
    __syncthreads();        // all waves' transfers visible; prev reads done
    if (k0 + GT_K < K) stage(cur ^ 1, k0 + GT_K);   // overlap next tile

#pragma unroll
    for (int ks = 0; ks < 2; ++ks) {
      const int kb = ks * 32;
      // A fragment: 16x32 bf16 (ISA layout: lane m = lane%16, half-wave K split)
      union Frag { v16bf v; unsigned u[8]; } af;
#pragma unroll
      for (int vv = 0; vv < 8; ++vv) {
        const int kk = kb + ((vv >> 2) << 4) + sel * 8 + ((vv & 3) << 1);
        af.u[vv] = *(const unsigned*)&As[cur][wave * 16 + mf][kk];
      }
#pragma unroll
      for (int nb = 0; nb < 4; ++nb) {
        union Frag bfv;
#pragma unroll
        for (int vv = 0; vv < 8; ++vv) {
          const int kk = kb + ((vv >> 2) << 4) + sel * 8 + ((vv & 3) << 1);
          bfv.u[vv] = *(const unsigned*)&Bs[cur][nb * 16 + mf][kk];
        }
        v8f* accp = (nb == 0) ? &acc0 : (nb == 1) ? &acc1 : (nb == 2) ? &acc2 : &acc3;
        *accp = __builtin_amdgcn_wmma_f32_16x16x32_bf16(
            false, af.v, false, bfv.v, (short)0, *accp, false, false);
      }
    }
    cur ^= 1;
  }

  // store: C/D layout — VGPR r, lanes<16 -> M=r, lanes>=16 -> M=r+8; N = lane%16
  union F8 { v8f v; float f[8]; } o[4];
  o[0].v = acc0; o[1].v = acc1; o[2].v = acc2; o[3].v = acc3;
#pragma unroll
  for (int nb = 0; nb < 4; ++nb) {
    const int col = tn + nb * 16 + mf;
    if (col >= N) continue;
    const float bv = bias ? bias[col] : 0.f;
#pragma unroll
    for (int r = 0; r < 8; ++r) {
      const int row = tm + wave * 16 + sel * 8 + r;
      if (row < M) {
        Y[(long)row * ldY + col] = f2b(act_apply(o[nb].f[r] + bv, act, actp));
      }
    }
  }
}

// ---------------------------------------------------------------- layout / cvt
// zero-padded weight pack: dst[Npad x Kpad] <- src[N x K]
__global__ void cvt_pad(const float* __restrict__ s, bfraw* __restrict__ d,
                        int N, int K, int Npad, int Kpad) {
  long i = (long)blockIdx.x * blockDim.x + threadIdx.x;
  if (i >= (long)Npad * Kpad) return;
  int k = (int)(i % Kpad); long n = i / Kpad;
  d[i] = (n < N && k < K) ? f2b(s[n * K + k]) : (bfraw)0;
}
__global__ void fill_zero_bf(bfraw* __restrict__ p, long n) {
  long i = (long)blockIdx.x * blockDim.x + threadIdx.x;
  if (i < n) p[i] = 0;
}
__global__ void nchw_to_tok(const float* __restrict__ x, bfraw* __restrict__ t) {
  long i = (long)blockIdx.x * blockDim.x + threadIdx.x;  // over B*C*HW (nchw order)
  if (i >= (long)BN * CC * HWSZ) return;
  int p = (int)(i % HWSZ); long r = i / HWSZ;
  int c = (int)(r % CC);   int b = (int)(r / CC);
  t[((long)b * HWSZ + p) * CC + c] = f2b(x[i]);
}
__global__ void tok_to_nchw_add(const bfraw* __restrict__ y, const bfraw* __restrict__ res,
                                float* __restrict__ out) {
  long i = (long)blockIdx.x * blockDim.x + threadIdx.x;
  if (i >= (long)BN * CC * HWSZ) return;
  int p = (int)(i % HWSZ); long r = i / HWSZ;
  int c = (int)(r % CC);   int b = (int)(r / CC);
  long t = ((long)b * HWSZ + p) * CC + c;
  out[i] = b2f(y[t]) + b2f(res[t]);
}

// ---------------------------------------------------------------- elementwise
__global__ void ew_add_act(const bfraw* a, const bfraw* b, bfraw* d, long n, int act, float p) {
  long i = (long)blockIdx.x * blockDim.x + threadIdx.x;
  if (i < n) d[i] = f2b(act_apply(b2f(a[i]) + b2f(b[i]), act, p));
}
__global__ void copy_cols(const bfraw* src, int ldS, bfraw* dst, int ldD, int colOff,
                          long rows, int cols) {
  long i = (long)blockIdx.x * blockDim.x + threadIdx.x;
  if (i >= rows * cols) return;
  int c = (int)(i % cols); long r = i / cols;
  dst[r * ldD + colOff + c] = src[r * ldS + c];
}
__global__ void scale_bc(bfraw* t, const float* s, int C, int HW, long n) {
  long i = (long)blockIdx.x * blockDim.x + threadIdx.x;
  if (i >= n) return;
  int c = (int)(i % C); long tok = i / C; int b = (int)(tok / HW);
  t[i] = f2b(b2f(t[i]) * s[b * C + c]);
}
__global__ void gelu_mul(const bfraw* u, bfraw* out, int ldU, int HID, int ldOut, long rows) {
  long i = (long)blockIdx.x * blockDim.x + threadIdx.x;
  if (i >= rows * HID) return;
  int c = (int)(i % HID); long r = i / HID;
  float a = b2f(u[r * ldU + c]);
  float g = 0.5f * a * (1.f + erff(a * 0.70710678f));
  out[r * ldOut + c] = f2b(g * b2f(u[r * ldU + HID + c]));
}

// ---------------------------------------------------------------- norms / reductions
__global__ __launch_bounds__(64) void layernorm_tok(const bfraw* __restrict__ src,
                                                    bfraw* __restrict__ dst,
                                                    const float* g, const float* bb, int C) {
  __shared__ float red[64];
  long tok = blockIdx.x;
  const bfraw* row = src + tok * C;
  bfraw* orow = dst + tok * C;
  int tid = threadIdx.x;
  float s = 0.f;
  for (int c = tid; c < C; c += 64) s += b2f(row[c]);
  red[tid] = s; __syncthreads();
  for (int k = 32; k > 0; k >>= 1) { if (tid < k) red[tid] += red[tid + k]; __syncthreads(); }
  float mu = red[0] / C; __syncthreads();
  float s2 = 0.f;
  for (int c = tid; c < C; c += 64) { float d = b2f(row[c]) - mu; s2 += d * d; }
  red[tid] = s2; __syncthreads();
  for (int k = 32; k > 0; k >>= 1) { if (tid < k) red[tid] += red[tid + k]; __syncthreads(); }
  float inv = rsqrtf(red[0] / C + 1e-5f);
  for (int c = tid; c < C; c += 64)
    orow[c] = f2b((b2f(row[c]) - mu) * inv * g[c] + bb[c]);
}
__global__ __launch_bounds__(256) void gap_tok(const bfraw* __restrict__ src, float* out,
                                               int C, int HW) {
  __shared__ float red[256];
  int bc = blockIdx.x; int b = bc / C; int c = bc % C;
  float s = 0.f;
  for (int t = threadIdx.x; t < HW; t += 256) s += b2f(src[((long)b * HW + t) * C + c]);
  red[threadIdx.x] = s; __syncthreads();
  for (int k = 128; k > 0; k >>= 1) { if (threadIdx.x < k) red[threadIdx.x] += red[threadIdx.x + k]; __syncthreads(); }
  if (threadIdx.x == 0) out[bc] = red[0] / HW;
}
__global__ void dense_small(const float* x, const float* w, const float* bias, float* y,
                            int Bn, int N, int K, int act) {
  int i = blockIdx.x * blockDim.x + threadIdx.x;
  if (i >= Bn * N) return;
  int b = i / N, n = i % N;
  float acc = bias ? bias[n] : 0.f;
  for (int k = 0; k < K; ++k) acc += x[b * K + k] * w[n * K + k];
  y[i] = act_apply(acc, act, 0.f);
}

// ---------------------------------------------------------------- depthwise convs
__global__ void dwconv3_tok(const bfraw* __restrict__ src, const float* __restrict__ w,
                            const float* __restrict__ bias, bfraw* __restrict__ dst,
                            int C, int ldC) {
  long i = (long)blockIdx.x * blockDim.x + threadIdx.x;
  if (i >= (long)NTOK * C) return;
  int c = (int)(i % C); long t = i / C;
  int x = (int)(t % WW); long t2 = t / WW;
  int y = (int)(t2 % HH); int b = (int)(t2 / HH);
  float acc = bias ? bias[c] : 0.f;
#pragma unroll
  for (int ky = 0; ky < 3; ++ky)
#pragma unroll
    for (int kx = 0; kx < 3; ++kx) {
      int yy = y + ky - 1, xx = x + kx - 1;
      if (yy >= 0 && yy < HH && xx >= 0 && xx < WW)
        acc += w[c * 9 + ky * 3 + kx] * b2f(src[(((long)b * HH + yy) * WW + xx) * ldC + c]);
    }
  dst[t * ldC + c] = f2b(acc);
}
__global__ void conv1d_silu(const bfraw* __restrict__ xz, const float* __restrict__ w,
                            const float* __restrict__ bias, bfraw* __restrict__ xc) {
  long i = (long)blockIdx.x * blockDim.x + threadIdx.x;  // over B*L*DI
  if (i >= (long)BN * HWSZ * DIC) return;
  int d = (int)(i % DIC); long bl = i / DIC;
  int l = (int)(bl % HWSZ); int b = (int)(bl / HWSZ);
  float acc = bias[d];
#pragma unroll
  for (int j = 0; j < 4; ++j) {
    int ls = l - 3 + j;
    if (ls >= 0) acc += w[d * 4 + j] * b2f(xz[((long)b * HWSZ + ls) * (2 * DIC) + d]);
  }
  acc = acc / (1.f + __expf(-acc));  // silu
  xc[i] = f2b(acc);
}

// ---------------------------------------------------------------- selective scan
__global__ void mamba_scan(const bfraw* __restrict__ xz, const bfraw* __restrict__ xc,
                           const bfraw* __restrict__ dt, const bfraw* __restrict__ xdbl,
                           const float* __restrict__ A_log, const float* __restrict__ Dv,
                           bfraw* __restrict__ y) {
  int id = blockIdx.x * blockDim.x + threadIdx.x;  // b*DI + d
  if (id >= BN * DIC) return;
  int b = id / DIC, d = id % DIC;
  float A[DSC], h[DSC];
#pragma unroll
  for (int s = 0; s < DSC; ++s) { A[s] = -__expf(A_log[d * DSC + s]); h[s] = 0.f; }
  float Dd = Dv[d];
  for (int l = 0; l < HWSZ; ++l) {
    long tok = (long)b * HWSZ + l;
    float dtv = b2f(dt[tok * DIC + d]);
    float xv  = b2f(xc[tok * DIC + d]);
    const bfraw* xr = &xdbl[tok * (DRC + 2 * DSC)];
    float accv = 0.f;
#pragma unroll
    for (int s = 0; s < DSC; ++s) {
      float Bv = b2f(xr[DRC + s]);
      float Cv = b2f(xr[DRC + DSC + s]);
      h[s] = __expf(dtv * A[s]) * h[s] + dtv * xv * Bv;
      accv += h[s] * Cv;
    }
    accv += xv * Dd;
    float z  = b2f(xz[tok * (2 * DIC) + DIC + d]);
    float sz = z / (1.f + __expf(-z));
    y[tok * DIC + d] = f2b(accv * sz);
  }
}

// ---------------------------------------------------------------- attention pieces
__global__ void tok_to_head(const bfraw* __restrict__ src, bfraw* __restrict__ dst,
                            int ld, int colOff) {
  long i = (long)blockIdx.x * blockDim.x + threadIdx.x;  // over B*256*HWSZ
  if (i >= (long)BN * CC * HWSZ) return;
  int n = (int)(i % HWSZ); long r = i / HWSZ;
  int ch = (int)(r % CC); int b = (int)(r / CC);
  dst[i] = src[((long)b * HWSZ + n) * ld + colOff + ch];
}
__global__ __launch_bounds__(256) void rownorm4096(bfraw* p) {
  __shared__ float red[256];
  long row = blockIdx.x;
  bfraw* r = p + row * HWSZ;
  float s = 0.f;
  for (int n = threadIdx.x; n < HWSZ; n += 256) { float v = b2f(r[n]); s += v * v; }
  red[threadIdx.x] = s; __syncthreads();
  for (int k = 128; k > 0; k >>= 1) { if (threadIdx.x < k) red[threadIdx.x] += red[threadIdx.x + k]; __syncthreads(); }
  float inv = 1.f / fmaxf(sqrtf(red[0]), 1e-12f);
  __syncthreads();
  for (int n = threadIdx.x; n < HWSZ; n += 256) r[n] = f2b(b2f(r[n]) * inv);
}
__global__ __launch_bounds__(64) void softmax64(const bfraw* __restrict__ attn,
                                                const float* __restrict__ temp,
                                                float* __restrict__ outp) {
  __shared__ float red[64];
  int idx = blockIdx.x;            // bh*64 + row
  int bh = idx >> 6;
  int head = bh & (NHEADS - 1);
  const bfraw* r = attn + (long)idx * 64;
  int tid = threadIdx.x;
  float v = b2f(r[tid]) * temp[head];
  red[tid] = v; __syncthreads();
  for (int k = 32; k > 0; k >>= 1) { if (tid < k) red[tid] = fmaxf(red[tid], red[tid + k]); __syncthreads(); }
  float mx = red[0]; __syncthreads();
  float e = __expf(v - mx);
  red[tid] = e; __syncthreads();
  for (int k = 32; k > 0; k >>= 1) { if (tid < k) red[tid] += red[tid + k]; __syncthreads(); }
  outp[(long)idx * 64 + tid] = e / red[0];
}
__global__ void apply_attn(const float* __restrict__ attnP, const bfraw* __restrict__ vh,
                           bfraw* __restrict__ out_tok) {
  long i = (long)blockIdx.x * blockDim.x + threadIdx.x;  // over 16*64*HWSZ
  if (i >= (long)BN * NHEADS * 64 * HWSZ) return;
  int n = (int)(i % HWSZ); long r = i / HWSZ;
  int cc = (int)(r & 63); int bh = (int)(r >> 6);
  int b = bh >> 2; int ch = (bh & 3) * 64 + cc;
  const float* arow = attnP + ((long)bh * 64 + cc) * 64;
  const bfraw* vb = vh + (long)bh * 64 * HWSZ + n;
  float acc = 0.f;
#pragma unroll 8
  for (int d = 0; d < 64; ++d) acc += arow[d] * b2f(vb[(long)d * HWSZ]);
  out_tok[((long)b * HWSZ + n) * CC + ch] = f2b(acc);
}

// ================================================================ host
extern "C" void kernel_launch(void* const* d_in, const int* in_sizes, int n_in,
                              void* d_out, int out_size, void* d_ws, size_t ws_size,
                              hipStream_t stream) {
  (void)in_sizes; (void)n_in; (void)out_size; (void)ws_size;

  int ii = 0;
  const float* x_in = (const float*)d_in[ii++];
  // afdu1
  const float *a1_ded_w = (const float*)d_in[ii++], *a1_ded_b = (const float*)d_in[ii++];
  const float *a1_exp_w = (const float*)d_in[ii++], *a1_exp_b = (const float*)d_in[ii++];
  const float *a1_c1_w  = (const float*)d_in[ii++], *a1_c1_b  = (const float*)d_in[ii++];
  const float *a1_ca1_w = (const float*)d_in[ii++], *a1_ca1_b = (const float*)d_in[ii++];
  const float *a1_ca2_w = (const float*)d_in[ii++], *a1_ca2_b = (const float*)d_in[ii++];
  const float *a1_c3d_w = (const float*)d_in[ii++], *a1_c3d_b = (const float*)d_in[ii++];
  const float *a1_c3p_w = (const float*)d_in[ii++], *a1_c3p_b = (const float*)d_in[ii++];
  // afdu2
  const float *a2_ded_w = (const float*)d_in[ii++], *a2_ded_b = (const float*)d_in[ii++];
  const float *a2_exp_w = (const float*)d_in[ii++], *a2_exp_b = (const float*)d_in[ii++];
  const float *a2_c1_w  = (const float*)d_in[ii++], *a2_c1_b  = (const float*)d_in[ii++];
  const float *a2_ca1_w = (const float*)d_in[ii++], *a2_ca1_b = (const float*)d_in[ii++];
  const float *a2_ca2_w = (const float*)d_in[ii++], *a2_ca2_b = (const float*)d_in[ii++];
  const float *a2_c3d_w = (const float*)d_in[ii++], *a2_c3d_b = (const float*)d_in[ii++];
  const float *a2_c3p_w = (const float*)d_in[ii++], *a2_c3p_b = (const float*)d_in[ii++];
  // mamba
  const float *ln_g = (const float*)d_in[ii++], *ln_b = (const float*)d_in[ii++];
  const float *in_w = (const float*)d_in[ii++];
  const float *conv_w = (const float*)d_in[ii++], *conv_b = (const float*)d_in[ii++];
  const float *xproj_w = (const float*)d_in[ii++];
  const float *dtproj_w = (const float*)d_in[ii++], *dtproj_b = (const float*)d_in[ii++];
  const float *A_log = (const float*)d_in[ii++], *Dvec = (const float*)d_in[ii++];
  const float *out_w = (const float*)d_in[ii++];
  const float *p1_w = (const float*)d_in[ii++], *p1_b = (const float*)d_in[ii++];
  const float *p2_w = (const float*)d_in[ii++], *p2_b = (const float*)d_in[ii++];
  // transformer
  const float *n1_g = (const float*)d_in[ii++], *n1_b = (const float*)d_in[ii++];
  const float *qkv_w = (const float*)d_in[ii++];
  const float *qkv_dw_w = (const float*)d_in[ii++];
  const float *po_w = (const float*)d_in[ii++];
  const float *temp = (const float*)d_in[ii++];
  const float *n2_g = (const float*)d_in[ii++], *n2_b = (const float*)d_in[ii++];
  const float *fi_w = (const float*)d_in[ii++];
  const float *fdw_w = (const float*)d_in[ii++];
  const float *fo_w = (const float*)d_in[ii++];

  // ---- workspace bump allocator
  char* wp = (char*)d_ws;
  auto alloc = [&](size_t bytes) -> void* {
    void* r = (void*)wp; wp += (bytes + 255) & ~(size_t)255; return r;
  };
  auto g1 = [&](long n) { return dim3((unsigned)((n + 255) / 256)); };

  // ---- zero-padded bf16 weight packs [Npad x Kpad]
  struct WCvt { const float* s; bfraw* d; int N, K, Npad, Kpad; };
  bfraw* wb_a1_ded = (bfraw*)alloc((long)128  * 256 * 2);
  bfraw* wb_a1_exp = (bfraw*)alloc((long)256  * 128 * 2);
  bfraw* wb_a1_c1  = (bfraw*)alloc((long)256  * 512 * 2);
  bfraw* wb_a1_c3p = (bfraw*)alloc((long)256  * 256 * 2);
  bfraw* wb_a2_ded = (bfraw*)alloc((long)128  * 256 * 2);
  bfraw* wb_a2_exp = (bfraw*)alloc((long)256  * 128 * 2);
  bfraw* wb_a2_c1  = (bfraw*)alloc((long)256  * 512 * 2);
  bfraw* wb_a2_c3p = (bfraw*)alloc((long)256  * 256 * 2);
  bfraw* wb_in   = (bfraw*)alloc((long)1024 * 256 * 2);
  bfraw* wb_xprj = (bfraw*)alloc((long)64   * 512 * 2);   // N 48 -> 64
  bfraw* wb_dtp  = (bfraw*)alloc((long)512  * 64  * 2);   // K 16 -> 64
  bfraw* wb_out  = (bfraw*)alloc((long)256  * 512 * 2);
  bfraw* wb_p1   = (bfraw*)alloc((long)512  * 256 * 2);
  bfraw* wb_p2   = (bfraw*)alloc((long)256  * 512 * 2);
  bfraw* wb_qkv  = (bfraw*)alloc((long)768  * 256 * 2);
  bfraw* wb_po   = (bfraw*)alloc((long)256  * 256 * 2);
  bfraw* wb_fi   = (bfraw*)alloc((long)1408 * 256 * 2);   // N 1360 -> 1408
  bfraw* wb_fo   = (bfraw*)alloc((long)256  * 704 * 2);   // K 680 -> 704

  const WCvt cvts[] = {
    {a1_ded_w, wb_a1_ded, 128, 256, 128, 256}, {a1_exp_w, wb_a1_exp, 256, 128, 256, 128},
    {a1_c1_w,  wb_a1_c1,  256, 512, 256, 512}, {a1_c3p_w, wb_a1_c3p, 256, 256, 256, 256},
    {a2_ded_w, wb_a2_ded, 128, 256, 128, 256}, {a2_exp_w, wb_a2_exp, 256, 128, 256, 128},
    {a2_c1_w,  wb_a2_c1,  256, 512, 256, 512}, {a2_c3p_w, wb_a2_c3p, 256, 256, 256, 256},
    {in_w, wb_in, 1024, 256, 1024, 256}, {xproj_w, wb_xprj, 48, 512, 64, 512},
    {dtproj_w, wb_dtp, 512, 16, 512, 64}, {out_w, wb_out, 256, 512, 256, 512},
    {p1_w, wb_p1, 512, 256, 512, 256}, {p2_w, wb_p2, 256, 512, 256, 512},
    {qkv_w, wb_qkv, 768, 256, 768, 256}, {po_w, wb_po, 256, 256, 256, 256},
    {fi_w, wb_fi, 1360, 256, 1408, 256}, {fo_w, wb_fo, 256, 680, 256, 704},
  };
  for (const auto& c : cvts)
    cvt_pad<<<g1((long)c.Npad * c.Kpad), 256, 0, stream>>>(c.s, c.d, c.N, c.K, c.Npad, c.Kpad);

  // ---- activations (bf16, token-major)
  const long TC = (long)NTOK * CC;
  bfraw* x_tok = (bfraw*)alloc(TC * 2);
  bfraw* thumb = (bfraw*)alloc((long)NTOK * 128 * 2);
  bfraw* cat   = (bfraw*)alloc((long)NTOK * 512 * 2);
  bfraw* c1b   = (bfraw*)alloc(TC * 2);
  bfraw* dwb   = (bfraw*)alloc(TC * 2);
  bfraw* c3o   = (bfraw*)alloc(TC * 2);
  float* pool  = (float*)alloc(BN * 256 * 4);
  float* h1    = (float*)alloc(BN * 128 * 4);
  float* sig   = (float*)alloc(BN * 256 * 4);
  bfraw* y1    = (bfraw*)alloc(TC * 2);
  bfraw* xn    = (bfraw*)alloc(TC * 2);
  bfraw* xzb   = (bfraw*)alloc((long)NTOK * 1024 * 2);
  bfraw* xcb   = (bfraw*)alloc((long)NTOK * 512 * 2);
  bfraw* xdbl  = (bfraw*)alloc((long)NTOK * 48 * 2);
  bfraw* dtin  = (bfraw*)alloc((long)NTOK * 64 * 2);      // dt cols zero-padded to K=64
  bfraw* dtb   = (bfraw*)alloc((long)NTOK * 512 * 2);
  bfraw* ysc   = (bfraw*)alloc((long)NTOK * 512 * 2);
  bfraw* mo    = (bfraw*)alloc(TC * 2);
  bfraw* xm    = (bfraw*)alloc(TC * 2);
  bfraw* xm2   = (bfraw*)alloc(TC * 2);
  bfraw* pb    = (bfraw*)alloc((long)NTOK * 512 * 2);
  bfraw* mlp   = (bfraw*)alloc(TC * 2);
  bfraw* idt2  = (bfraw*)alloc(TC * 2);
  bfraw* tt    = (bfraw*)alloc(TC * 2);
  bfraw* n1bf  = (bfraw*)alloc(TC * 2);
  bfraw* qkvb  = (bfraw*)alloc((long)NTOK * 768 * 2);
  bfraw* qkvd  = (bfraw*)alloc((long)NTOK * 768 * 2);
  bfraw* qh    = (bfraw*)alloc(TC * 2);   // [16][64][4096]
  bfraw* kh    = (bfraw*)alloc(TC * 2);
  bfraw* vh    = (bfraw*)alloc(TC * 2);
  bfraw* attnB = (bfraw*)alloc((long)16 * 64 * 64 * 2);
  float* attnP = (float*)alloc((long)16 * 64 * 64 * 4);
  bfraw* atok  = (bfraw*)alloc(TC * 2);
  bfraw* poo   = (bfraw*)alloc(TC * 2);
  bfraw* t2    = (bfraw*)alloc(TC * 2);
  bfraw* n2bf  = (bfraw*)alloc(TC * 2);
  bfraw* fib   = (bfraw*)alloc((long)NTOK * 1408 * 2);
  bfraw* fid   = (bfraw*)alloc((long)NTOK * 1408 * 2);
  bfraw* gat   = (bfraw*)alloc((long)NTOK * 704 * 2);     // ld 704, pad zeroed
  bfraw* foo   = (bfraw*)alloc(TC * 2);
  bfraw* t3    = (bfraw*)alloc(TC * 2);

  auto gemm = [&](const bfraw* A, const bfraw* Wt, const float* bias, bfraw* Yp,
                  int M, int N, int K, int ldA, int Kw, int ldY, int act, float actp,
                  int batches = 1, long sA = 0, long sW = 0, long sY = 0) {
    dim3 grid((N + GT_N - 1) / GT_N, (M + GT_M - 1) / GT_M, batches);
    gemm_bf16_wmma<<<grid, 256, 0, stream>>>(A, Wt, bias, Yp, M, N, K, ldA, Kw, ldY,
                                             act, actp, sA, sW, sY);
  };

  auto afdu = [&](const bfraw* f, bfraw* out,
                  const bfraw* dedW, const float* dedB, const bfraw* expW, const float* expB,
                  const bfraw* c1W, const float* c1B,
                  const float* ca1W, const float* ca1B, const float* ca2W, const float* ca2B,
                  const float* c3dW, const float* c3dB, const bfraw* c3pW, const float* c3pB) {
    gemm(f, dedW, dedB, thumb, NTOK, 128, 256, 256, 256, 128, 1, 0.01f);   // thumb = leaky
    gemm(thumb, expW, expB, cat, NTOK, 256, 128, 128, 128, 512, 0, 0.f);   // exp -> cat[:, :256]
    copy_cols<<<g1(TC), 256, 0, stream>>>(f, 256, cat, 512, 256, NTOK, 256);
    gemm(cat, c1W, c1B, c1b, NTOK, 256, 512, 512, 512, 256, 1, 0.01f);     // c1 = leaky
    gap_tok<<<BN * CC, 256, 0, stream>>>(c1b, pool, CC, HWSZ);
    dense_small<<<g1(BN * 128), 256, 0, stream>>>(pool, ca1W, ca1B, h1, BN, 128, 256, 2);
    dense_small<<<g1(BN * 256), 256, 0, stream>>>(h1, ca2W, ca2B, sig, BN, 256, 128, 3);
    scale_bc<<<g1(TC), 256, 0, stream>>>(c1b, sig, CC, HWSZ, TC);          // att = c1 * y
    dwconv3_tok<<<g1(TC), 256, 0, stream>>>(c1b, c3dW, c3dB, dwb, CC, CC);
    gemm(dwb, c3pW, c3pB, c3o, NTOK, 256, 256, 256, 256, 256, 0, 0.f);
    ew_add_act<<<g1(TC), 256, 0, stream>>>(c3o, f, out, TC, 1, 0.2f);      // leaky(c3+f, 0.2)
  };

  // ---- layout input
  nchw_to_tok<<<g1((long)BN * CC * HWSZ), 256, 0, stream>>>(x_in, x_tok);

  // ---- AFDU 1 (+ fused outer leaky 0.2)
  afdu(x_tok, y1, wb_a1_ded, a1_ded_b, wb_a1_exp, a1_exp_b, wb_a1_c1, a1_c1_b,
       a1_ca1_w, a1_ca1_b, a1_ca2_w, a1_ca2_b, a1_c3d_w, a1_c3d_b, wb_a1_c3p, a1_c3p_b);

  // ---- Mamba block
  layernorm_tok<<<NTOK, 64, 0, stream>>>(y1, xn, ln_g, ln_b, CC);
  gemm(xn, wb_in, nullptr, xzb, NTOK, 1024, 256, 256, 256, 1024, 0, 0.f);  // xz
  conv1d_silu<<<g1((long)NTOK * DIC), 256, 0, stream>>>(xzb, conv_w, conv_b, xcb);
  gemm(xcb, wb_xprj, nullptr, xdbl, NTOK, 48, 512, 512, 512, 48, 0, 0.f);  // dt|B|C
  fill_zero_bf<<<g1((long)NTOK * 64), 256, 0, stream>>>(dtin, (long)NTOK * 64);
  copy_cols<<<g1((long)NTOK * 16), 256, 0, stream>>>(xdbl, 48, dtin, 64, 0, NTOK, 16);
  gemm(dtin, wb_dtp, dtproj_b, dtb, NTOK, 512, 64, 64, 64, 512, 6, 0.f);   // softplus
  mamba_scan<<<(BN * DIC + 255) / 256, 256, 0, stream>>>(xzb, xcb, dtb, xdbl, A_log, Dvec, ysc);
  gemm(ysc, wb_out, nullptr, mo, NTOK, 256, 512, 512, 512, 256, 0, 0.f);
  ew_add_act<<<g1(TC), 256, 0, stream>>>(mo, y1, xm, TC, 0, 0.f);          // + xe
  layernorm_tok<<<NTOK, 64, 0, stream>>>(xm, xm2, ln_g, ln_b, CC);
  gemm(xm2, wb_p1, p1_b, pb, NTOK, 512, 256, 256, 256, 512, 5, 0.f);       // gelu
  gemm(pb, wb_p2, p2_b, mlp, NTOK, 256, 512, 512, 512, 256, 0, 0.f);
  ew_add_act<<<g1(TC), 256, 0, stream>>>(mlp, x_tok, idt2, TC, 0, 0.f);    // + idt

  // ---- AFDU 2 (+ fused outer leaky 0.2)
  afdu(idt2, tt, wb_a2_ded, a2_ded_b, wb_a2_exp, a2_exp_b, wb_a2_c1, a2_c1_b,
       a2_ca1_w, a2_ca1_b, a2_ca2_w, a2_ca2_b, a2_c3d_w, a2_c3d_b, wb_a2_c3p, a2_c3p_b);

  // ---- Transformer: attention
  layernorm_tok<<<NTOK, 64, 0, stream>>>(tt, n1bf, n1_g, n1_b, CC);
  gemm(n1bf, wb_qkv, nullptr, qkvb, NTOK, 768, 256, 256, 256, 768, 0, 0.f);
  dwconv3_tok<<<g1((long)NTOK * 768), 256, 0, stream>>>(qkvb, qkv_dw_w, nullptr, qkvd, 768, 768);
  tok_to_head<<<g1((long)BN * CC * HWSZ), 256, 0, stream>>>(qkvd, qh, 768, 0);
  tok_to_head<<<g1((long)BN * CC * HWSZ), 256, 0, stream>>>(qkvd, kh, 768, 256);
  tok_to_head<<<g1((long)BN * CC * HWSZ), 256, 0, stream>>>(qkvd, vh, 768, 512);
  rownorm4096<<<BN * CC, 256, 0, stream>>>(qh);
  rownorm4096<<<BN * CC, 256, 0, stream>>>(kh);
  // channel-gram per (b,head): attn[64,64] = q(64,4096) . k(64,4096)^T  — batched WMMA
  gemm(qh, kh, nullptr, attnB, 64, 64, HWSZ, HWSZ, HWSZ, 64, 0, 0.f,
       BN * NHEADS, (long)64 * HWSZ, (long)64 * HWSZ, (long)64 * 64);
  softmax64<<<BN * NHEADS * 64, 64, 0, stream>>>(attnB, temp, attnP);
  apply_attn<<<g1((long)BN * NHEADS * 64 * HWSZ), 256, 0, stream>>>(attnP, vh, atok);
  gemm(atok, wb_po, nullptr, poo, NTOK, 256, 256, 256, 256, 256, 0, 0.f);
  ew_add_act<<<g1(TC), 256, 0, stream>>>(tt, poo, t2, TC, 0, 0.f);

  // ---- Transformer: FFN
  layernorm_tok<<<NTOK, 64, 0, stream>>>(t2, n2bf, n2_g, n2_b, CC);
  gemm(n2bf, wb_fi, nullptr, fib, NTOK, 1360, 256, 256, 256, 1408, 0, 0.f);
  dwconv3_tok<<<g1((long)NTOK * 1360), 256, 0, stream>>>(fib, fdw_w, nullptr, fid, 1360, 1408);
  fill_zero_bf<<<g1((long)NTOK * 704), 256, 0, stream>>>(gat, (long)NTOK * 704);
  gelu_mul<<<g1((long)NTOK * HIDC), 256, 0, stream>>>(fid, gat, 1408, HIDC, 704, NTOK);
  gemm(gat, wb_fo, nullptr, foo, NTOK, 256, 704, 704, 704, 256, 0, 0.f);
  ew_add_act<<<g1(TC), 256, 0, stream>>>(t2, foo, t3, TC, 0, 0.f);

  // ---- final residual + NCHW f32 output
  tok_to_nchw_add<<<g1((long)BN * CC * HWSZ), 256, 0, stream>>>(t3, idt2, (float*)d_out);
}
